// GPTModel_48687749267792
// MI455X (gfx1250) — compile-verified, matrix operand order checked by reading, not run on hardware
//
#include <hip/hip_runtime.h>
#include <math.h>

// ---------------- model constants (match reference) ----------------
#define D_MODEL 1024
#define T_SEQ   1024
#define B_BATCH 2
#define N_HEAD  16
#define HD_DIM  64
#define N_LAYER 8
#define VOCAB   32000
#define ROWS_M  (B_BATCH * T_SEQ)   // 2048
#define ATT_SCALE 0.125f            // 1/sqrt(64)

// ---------------- WMMA plumbing (CDNA5 / gfx1250, wave32) ----------
typedef __attribute__((ext_vector_type(16))) __bf16 v16bf;
typedef __attribute__((ext_vector_type(8)))  float  v8f;
typedef __attribute__((ext_vector_type(4)))  unsigned int u32x4;

union FragU { u32x4 u[2]; v16bf v; };

__device__ __forceinline__ unsigned short f2bf(float f) {
  union { float f; unsigned u; } c; c.f = f;
  unsigned r = c.u + 0x7FFFu + ((c.u >> 16) & 1u);   // round-to-nearest-even
  return (unsigned short)(r >> 16);
}

__device__ __forceinline__ unsigned pack2bf(float a, float b) {
  return (unsigned)f2bf(a) | ((unsigned)f2bf(b) << 16);
}

__device__ __forceinline__ v8f wmma_bf16(v16bf a, v16bf b, v8f c) {
  // (neg_a, A, neg_b, B, c_mod, C, reuse_a, reuse_b)
  return __builtin_amdgcn_wmma_f32_16x16x32_bf16(false, a, false, b, (short)0, c,
                                                 false, false);
}

// Load one 16x32 bf16 A/B fragment from LDS. `base` points at the tile's
// [16][stride] sub-block (rows = the 16-dim; K contiguous within a row).
// ISA layout (16-bit A 16x32): lane = half*16 + m; element e holds
// K = (e<8 ? e : e+8) + half*8  -> two contiguous 8-bf16 (16B) runs per lane.
// B fragments mirror this (lane = half*16 + n), so [N][K] LDS tiles serve both.
__device__ __forceinline__ v16bf load_frag(const unsigned short* base, int lane,
                                           int stride /* ushorts, mult of 8 */) {
  int idx = (lane & 15) * stride + (lane >> 4) * 8;
  FragU f;
  f.u[0] = *(const u32x4*)(base + idx);        // K = half*8 + [0..7]
  f.u[1] = *(const u32x4*)(base + idx + 16);   // K = 16 + half*8 + [0..7]
  return f.v;
}

// ---------------- embed: x = tok_emb[ids] + pos_emb ----------------
__global__ __launch_bounds__(256) void embed_kernel(
    const int* __restrict__ ids, const float* __restrict__ tok,
    const float* __restrict__ pos, float* __restrict__ x) {
  size_t i = (size_t)blockIdx.x * blockDim.x + threadIdx.x;
  size_t total = (size_t)ROWS_M * D_MODEL;
  if (i >= total) return;
  size_t t = i / D_MODEL, d = i % D_MODEL;
  size_t tp = t % T_SEQ;
  x[i] = tok[(size_t)ids[t] * D_MODEL + d] + pos[tp * D_MODEL + d];
}

// ---------------- layernorm: one 256-thread block per row ----------
__global__ __launch_bounds__(256) void layernorm_kernel(
    const float* __restrict__ x, const float* __restrict__ w,
    const float* __restrict__ b, float* __restrict__ y) {
  __shared__ float red[256];
  const int tid = threadIdx.x;
  const float* xr = x + (size_t)blockIdx.x * D_MODEL;
  float4 v4 = *(const float4*)(xr + tid * 4);
  float v[4] = {v4.x, v4.y, v4.z, v4.w};
  float s = v[0] + v[1] + v[2] + v[3];
  red[tid] = s; __syncthreads();
  for (int off = 128; off > 0; off >>= 1) {
    if (tid < off) red[tid] += red[tid + off];
    __syncthreads();
  }
  float mean = red[0] * (1.0f / D_MODEL);
  __syncthreads();
  float sq = 0.f;
  #pragma unroll
  for (int i = 0; i < 4; ++i) { float d = v[i] - mean; sq += d * d; }
  red[tid] = sq; __syncthreads();
  for (int off = 128; off > 0; off >>= 1) {
    if (tid < off) red[tid] += red[tid + off];
    __syncthreads();
  }
  float inv = rsqrtf(red[0] * (1.0f / D_MODEL) + 1e-5f);
  float* yr = y + (size_t)blockIdx.x * D_MODEL;
  #pragma unroll
  for (int i = 0; i < 4; ++i) {
    int idx = tid * 4 + i;
    yr[idx] = (v[i] - mean) * inv * w[idx] + b[idx];
  }
}

// ---------------- tiled WMMA GEMM (128x128x32, double-buffered) -----
// C[M][N] = epilogue( A[M][K] @ B + bias ) (+ residual)
// TRANS_B=0: B is [K][N] row-major; TRANS_B=1: B is [N][K] row-major.
// Requires M%128==0, N%128==0, K%32==0 (true for all uses here).
#define BM 128
#define BN 128
#define BK 32
#define BKP 40   // padded K stride (ushorts) to stagger LDS banks

template <bool TRANS_B, bool RESID, bool GELU>
__global__ __launch_bounds__(256) void gemm_kernel(
    const float* __restrict__ A, const float* __restrict__ B,
    const float* __restrict__ bias, const float* __restrict__ res,
    float* __restrict__ C, int M, int N, int K) {
  __shared__ __attribute__((aligned(16))) unsigned short As[2 * BM * BKP];
  __shared__ __attribute__((aligned(16))) unsigned short Bs[2 * BN * BKP];
  const int tid = threadIdx.x, lane = tid & 31, wid = tid >> 5;
  const int wm = wid & 1, wn = wid >> 1;          // 2x4 waves of 64x32
  const int m0 = blockIdx.y * BM, n0 = blockIdx.x * BN;

  v8f acc[4][2] = {};
  float4 ar[4], br[4];
  const int nk = K / BK;

  // staging index precompute
  const int a_kq = (tid & 7) * 4, a_mr = tid >> 3;          // A & trans-B pattern
  const int b_n4 = (tid & 31) * 4, b_kk = tid >> 5;         // non-trans B pattern

  // ---- prologue: global-load tile 0 ----
  #pragma unroll
  for (int r = 0; r < 4; ++r)
    ar[r] = *(const float4*)(A + (size_t)(m0 + a_mr + r * 32) * K + a_kq);
  if (TRANS_B) {
    #pragma unroll
    for (int r = 0; r < 4; ++r)
      br[r] = *(const float4*)(B + (size_t)(n0 + a_mr + r * 32) * K + a_kq);
  } else {
    #pragma unroll
    for (int r = 0; r < 4; ++r)
      br[r] = *(const float4*)(B + (size_t)(b_kk + r * 8) * N + n0 + b_n4);
  }
  // ---- store tile 0 into buffer 0 ----
  {
    unsigned short* Asb = As;
    unsigned short* Bsb = Bs;
    #pragma unroll
    for (int r = 0; r < 4; ++r) {
      uint2 p = {pack2bf(ar[r].x, ar[r].y), pack2bf(ar[r].z, ar[r].w)};
      *(uint2*)(&Asb[(a_mr + r * 32) * BKP + a_kq]) = p;
    }
    if (TRANS_B) {
      #pragma unroll
      for (int r = 0; r < 4; ++r) {
        uint2 p = {pack2bf(br[r].x, br[r].y), pack2bf(br[r].z, br[r].w)};
        *(uint2*)(&Bsb[(a_mr + r * 32) * BKP + a_kq]) = p;
      }
    } else {
      #pragma unroll
      for (int r = 0; r < 4; ++r) {
        int k = b_kk + r * 8;
        Bsb[(b_n4 + 0) * BKP + k] = f2bf(br[r].x);
        Bsb[(b_n4 + 1) * BKP + k] = f2bf(br[r].y);
        Bsb[(b_n4 + 2) * BKP + k] = f2bf(br[r].z);
        Bsb[(b_n4 + 3) * BKP + k] = f2bf(br[r].w);
      }
    }
  }
  __syncthreads();

  for (int ki = 0; ki < nk; ++ki) {
    const int cur = ki & 1, nxt = cur ^ 1;
    const bool more = (ki + 1) < nk;
    // ---- issue global loads for next tile (latency hidden under WMMAs) ----
    if (more) {
      const int k0n = (ki + 1) * BK;
      #pragma unroll
      for (int r = 0; r < 4; ++r)
        ar[r] = *(const float4*)(A + (size_t)(m0 + a_mr + r * 32) * K + k0n + a_kq);
      if (TRANS_B) {
        #pragma unroll
        for (int r = 0; r < 4; ++r)
          br[r] = *(const float4*)(B + (size_t)(n0 + a_mr + r * 32) * K + k0n + a_kq);
      } else {
        #pragma unroll
        for (int r = 0; r < 4; ++r)
          br[r] = *(const float4*)(B + (size_t)(k0n + b_kk + r * 8) * N + n0 + b_n4);
      }
    }
    if (ki + 2 < nk) {   // prefetch 2 tiles ahead into cache (global_prefetch_b8)
      const int k0p = (ki + 2) * BK;
      __builtin_prefetch(A + (size_t)(m0 + a_mr) * K + k0p + a_kq, 0, 1);
      if (TRANS_B)
        __builtin_prefetch(B + (size_t)(n0 + a_mr) * K + k0p + a_kq, 0, 1);
      else
        __builtin_prefetch(B + (size_t)(k0p + b_kk) * N + n0 + b_n4, 0, 1);
    }
    // ---- 8 WMMAs per wave from current buffer ----
    {
      const unsigned short* Ac = As + cur * (BM * BKP);
      const unsigned short* Bc = Bs + cur * (BN * BKP);
      v16bf af[4], bfr[2];
      #pragma unroll
      for (int sm = 0; sm < 4; ++sm)
        af[sm] = load_frag(&Ac[(wm * 64 + sm * 16) * BKP], lane, BKP);
      #pragma unroll
      for (int sn = 0; sn < 2; ++sn)
        bfr[sn] = load_frag(&Bc[(wn * 32 + sn * 16) * BKP], lane, BKP);
      #pragma unroll
      for (int sm = 0; sm < 4; ++sm)
        #pragma unroll
        for (int sn = 0; sn < 2; ++sn)
          acc[sm][sn] = wmma_bf16(af[sm], bfr[sn], acc[sm][sn]);
    }
    // ---- store staged tile into other buffer ----
    if (more) {
      unsigned short* Asb = As + nxt * (BM * BKP);
      unsigned short* Bsb = Bs + nxt * (BN * BKP);
      #pragma unroll
      for (int r = 0; r < 4; ++r) {
        uint2 p = {pack2bf(ar[r].x, ar[r].y), pack2bf(ar[r].z, ar[r].w)};
        *(uint2*)(&Asb[(a_mr + r * 32) * BKP + a_kq]) = p;
      }
      if (TRANS_B) {
        #pragma unroll
        for (int r = 0; r < 4; ++r) {
          uint2 p = {pack2bf(br[r].x, br[r].y), pack2bf(br[r].z, br[r].w)};
          *(uint2*)(&Bsb[(a_mr + r * 32) * BKP + a_kq]) = p;
        }
      } else {
        #pragma unroll
        for (int r = 0; r < 4; ++r) {
          int k = b_kk + r * 8;
          Bsb[(b_n4 + 0) * BKP + k] = f2bf(br[r].x);
          Bsb[(b_n4 + 1) * BKP + k] = f2bf(br[r].y);
          Bsb[(b_n4 + 2) * BKP + k] = f2bf(br[r].z);
          Bsb[(b_n4 + 3) * BKP + k] = f2bf(br[r].w);
        }
      }
    }
    __syncthreads();
  }

  // --- epilogue: C layout row = v + (lane>>4)*8, col = lane&15 ---
  const int ln = lane & 15, half = lane >> 4;
  #pragma unroll
  for (int sm = 0; sm < 4; ++sm)
    #pragma unroll
    for (int sn = 0; sn < 2; ++sn) {
      int colg = n0 + wn * 32 + sn * 16 + ln;
      float bv = bias ? bias[colg] : 0.f;
      #pragma unroll
      for (int v = 0; v < 8; ++v) {
        int rowg = m0 + wm * 64 + sm * 16 + v + half * 8;
        float val = acc[sm][sn][v] + bv;
        if (GELU) val = 0.5f * val * (1.0f + erff(val * 0.70710678118654752f));
        if (RESID) val += res[(size_t)rowg * N + colg];
        C[(size_t)rowg * N + colg] = val;
      }
    }
}

// ---------------- fused causal flash attention ---------------------
// grid = (T/64, H, B), 128 threads (4 waves); wave w owns q-rows [w*16, w*16+16)
#define HDP 72   // padded 64 stride (ushorts)
#define SKP 68   // padded 64 stride (floats)
__global__ __launch_bounds__(128) void attention_kernel(
    const float* __restrict__ qkv, float* __restrict__ o) {
  __shared__ __attribute__((aligned(16))) unsigned short Qs[64 * HDP];
  __shared__ __attribute__((aligned(16))) unsigned short Ks[64 * HDP];
  __shared__ __attribute__((aligned(16))) unsigned short Vt[64 * HDP];
  __shared__ __attribute__((aligned(16))) unsigned short Ps[64 * HDP];
  __shared__ float Sf[64 * SKP];
  __shared__ float mrow[64], lrow[64], arow[64];

  const int tid = threadIdx.x, lane = tid & 31, wid = tid >> 5;
  const int q0 = blockIdx.x * 64, h = blockIdx.y, b = blockIdx.z;
  const size_t strideT = 3 * D_MODEL;
  const size_t baseQ = h * HD_DIM;
  const size_t baseK = D_MODEL + h * HD_DIM;
  const size_t baseV = 2 * D_MODEL + h * HD_DIM;

  // stage Q tile 64x64 -> bf16
  {
    int hd4 = (tid & 15) * 4, pr = tid >> 4;
    #pragma unroll
    for (int r = 0; r < 8; ++r) {
      int row = pr + r * 8;
      float4 v = *(const float4*)(qkv + (size_t)(b * T_SEQ + q0 + row) * strideT + baseQ + hd4);
      uint2 p = {pack2bf(v.x, v.y), pack2bf(v.z, v.w)};
      *(uint2*)(&Qs[row * HDP + hd4]) = p;
    }
  }
  if (tid < 64) { mrow[tid] = -3.0e38f; lrow[tid] = 0.f; }
  __syncthreads();

  v16bf qf0 = load_frag(&Qs[(wid * 16) * HDP +  0], lane, HDP);
  v16bf qf1 = load_frag(&Qs[(wid * 16) * HDP + 32], lane, HDP);
  v8f Oa[4] = {};

  const int ln = lane & 15, half = lane >> 4;

  for (int kb = 0; kb < q0 + 64; kb += 64) {
    // stage K block [pos][hd] and V block transposed [hd][pos]
    {
      int hd4 = (tid & 15) * 4, pr = tid >> 4;
      #pragma unroll
      for (int r = 0; r < 8; ++r) {
        int p = pr + r * 8;
        size_t rb = (size_t)(b * T_SEQ + kb + p) * strideT;
        float4 kv = *(const float4*)(qkv + rb + baseK + hd4);
        uint2 pk = {pack2bf(kv.x, kv.y), pack2bf(kv.z, kv.w)};
        *(uint2*)(&Ks[p * HDP + hd4]) = pk;
        float4 vv = *(const float4*)(qkv + rb + baseV + hd4);
        Vt[(hd4 + 0) * HDP + p] = f2bf(vv.x);
        Vt[(hd4 + 1) * HDP + p] = f2bf(vv.y);
        Vt[(hd4 + 2) * HDP + p] = f2bf(vv.z);
        Vt[(hd4 + 3) * HDP + p] = f2bf(vv.w);
      }
    }
    __syncthreads();
    // S = Q @ K^T * scale, causal mask -> LDS (fp32)
    #pragma unroll
    for (int sn = 0; sn < 4; ++sn) {
      v8f s = {};
      v16bf k0f = load_frag(&Ks[(sn * 16) * HDP +  0], lane, HDP);
      v16bf k1f = load_frag(&Ks[(sn * 16) * HDP + 32], lane, HDP);
      s = wmma_bf16(qf0, k0f, s);
      s = wmma_bf16(qf1, k1f, s);
      #pragma unroll
      for (int v = 0; v < 8; ++v) {
        int lr = wid * 16 + v + half * 8;
        int kpos = kb + sn * 16 + ln;
        float val = (kpos <= q0 + lr) ? s[v] * ATT_SCALE : -3.0e38f;
        Sf[lr * SKP + sn * 16 + ln] = val;
      }
    }
    __syncthreads();
    // online softmax bookkeeping, one thread per q-row
    if (tid < 64) {
      int r = tid;
      float bm = -3.0e38f;
      for (int j = 0; j < 64; ++j) bm = fmaxf(bm, Sf[r * SKP + j]);
      float mnew = fmaxf(mrow[r], bm);
      float alpha = __expf(mrow[r] - mnew);
      float sum = 0.f;
      for (int j = 0; j < 64; ++j) {
        float p = __expf(Sf[r * SKP + j] - mnew);
        Ps[r * HDP + j] = f2bf(p);
        sum += p;
      }
      lrow[r] = lrow[r] * alpha + sum;
      mrow[r] = mnew;
      arow[r] = alpha;
    }
    __syncthreads();
    // O = O*alpha + P @ V
    {
      float al[8];
      #pragma unroll
      for (int v = 0; v < 8; ++v) al[v] = arow[wid * 16 + v + half * 8];
      #pragma unroll
      for (int sn = 0; sn < 4; ++sn)
        #pragma unroll
        for (int v = 0; v < 8; ++v) Oa[sn][v] *= al[v];
      v16bf p0 = load_frag(&Ps[(wid * 16) * HDP +  0], lane, HDP);
      v16bf p1 = load_frag(&Ps[(wid * 16) * HDP + 32], lane, HDP);
      #pragma unroll
      for (int sn = 0; sn < 4; ++sn) {
        v16bf v0 = load_frag(&Vt[(sn * 16) * HDP +  0], lane, HDP);
        v16bf v1 = load_frag(&Vt[(sn * 16) * HDP + 32], lane, HDP);
        Oa[sn] = wmma_bf16(p0, v0, Oa[sn]);
        Oa[sn] = wmma_bf16(p1, v1, Oa[sn]);
      }
    }
    __syncthreads();
  }

  // write O / l  -> o[b, q, h*64 + hd]
  #pragma unroll
  for (int sn = 0; sn < 4; ++sn)
    #pragma unroll
    for (int v = 0; v < 8; ++v) {
      int lr = wid * 16 + v + half * 8;
      float invl = 1.0f / lrow[lr];
      int q = q0 + lr, hd = sn * 16 + ln;
      o[(size_t)(b * T_SEQ + q) * D_MODEL + h * HD_DIM + hd] = Oa[sn][v] * invl;
    }
}

// ---------------- driver -------------------------------------------
extern "C" void kernel_launch(void* const* d_in, const int* in_sizes, int n_in,
                              void* d_out, int out_size, void* d_ws, size_t ws_size,
                              hipStream_t stream) {
  (void)in_sizes; (void)n_in; (void)out_size; (void)ws_size;
  const int*   ids    = (const int*)  d_in[0];
  const float* tok    = (const float*)d_in[1];
  const float* pos    = (const float*)d_in[2];
  const float* ln1_w  = (const float*)d_in[3];
  const float* ln1_b  = (const float*)d_in[4];
  const float* qkv_w  = (const float*)d_in[5];
  const float* out_w  = (const float*)d_in[6];
  const float* out_b  = (const float*)d_in[7];
  const float* ln2_w  = (const float*)d_in[8];
  const float* ln2_b  = (const float*)d_in[9];
  const float* fc1_w  = (const float*)d_in[10];
  const float* fc1_b  = (const float*)d_in[11];
  const float* fc2_w  = (const float*)d_in[12];
  const float* fc2_b  = (const float*)d_in[13];
  const float* lnf_w  = (const float*)d_in[14];
  const float* lnf_b  = (const float*)d_in[15];

  const size_t ND = (size_t)ROWS_M * D_MODEL;       // 2,097,152
  float* x  = (float*)d_ws;                         // [B*T, D]  residual stream
  float* h  = x + ND;                               // [B*T, D]  ln output
  float* ob = h + ND;                               // [B*T, D]  attention output
  // big intermediates staged inside d_out (fully overwritten by lm_head last)
  float* qkvb = (float*)d_out;                      // [B*T, 3D]
  float* f1   = qkvb + (size_t)ROWS_M * 3 * D_MODEL;// [B*T, 4D]

  embed_kernel<<<dim3((unsigned)((ND + 255) / 256)), 256, 0, stream>>>(ids, tok, pos, x);

  for (int l = 0; l < N_LAYER; ++l) {
    const float* l1w = ln1_w + (size_t)l * D_MODEL;
    const float* l1b = ln1_b + (size_t)l * D_MODEL;
    const float* qw  = qkv_w + (size_t)l * D_MODEL * 3 * D_MODEL;
    const float* ow  = out_w + (size_t)l * D_MODEL * D_MODEL;
    const float* obv = out_b + (size_t)l * D_MODEL;
    const float* l2w = ln2_w + (size_t)l * D_MODEL;
    const float* l2b = ln2_b + (size_t)l * D_MODEL;
    const float* f1w = fc1_w + (size_t)l * D_MODEL * 4 * D_MODEL;
    const float* f1b = fc1_b + (size_t)l * 4 * D_MODEL;
    const float* f2w = fc2_w + (size_t)l * 4 * D_MODEL * D_MODEL;
    const float* f2b = fc2_b + (size_t)l * D_MODEL;

    layernorm_kernel<<<ROWS_M, 256, 0, stream>>>(x, l1w, l1b, h);
    gemm_kernel<false, false, false><<<dim3(3 * D_MODEL / BN, ROWS_M / BM), 256, 0, stream>>>(
        h, qw, nullptr, nullptr, qkvb, ROWS_M, 3 * D_MODEL, D_MODEL);
    attention_kernel<<<dim3(T_SEQ / 64, N_HEAD, B_BATCH), 128, 0, stream>>>(qkvb, ob);
    // x = x + o @ out_w + out_b  (in place: one thread per element)
    gemm_kernel<false, true, false><<<dim3(D_MODEL / BN, ROWS_M / BM), 256, 0, stream>>>(
        ob, ow, obv, x, x, ROWS_M, D_MODEL, D_MODEL);
    layernorm_kernel<<<ROWS_M, 256, 0, stream>>>(x, l2w, l2b, h);
    // f1 = gelu(h @ fc1_w + fc1_b)
    gemm_kernel<false, false, true><<<dim3(4 * D_MODEL / BN, ROWS_M / BM), 256, 0, stream>>>(
        h, f1w, f1b, nullptr, f1, ROWS_M, 4 * D_MODEL, D_MODEL);
    // x = x + f1 @ fc2_w + fc2_b
    gemm_kernel<false, true, false><<<dim3(D_MODEL / BN, ROWS_M / BM), 256, 0, stream>>>(
        f1, f2w, f2b, x, x, ROWS_M, D_MODEL, 4 * D_MODEL);
  }

  layernorm_kernel<<<ROWS_M, 256, 0, stream>>>(x, lnf_w, lnf_b, h);
  // logits = h @ tok_emb^T  (tok_emb is [V, D] row-major == B[N][K])
  gemm_kernel<true, false, false><<<dim3(VOCAB / BN, ROWS_M / BM), 256, 0, stream>>>(
      h, tok, nullptr, nullptr, (float*)d_out, ROWS_M, VOCAB, D_MODEL);
}
